// Converter_21904333209877
// MI455X (gfx1250) — matrix operand order repaired and checked by reading, not compiled, not run on hardware
//
#include <hip/hip_runtime.h>

// ---------------------------------------------------------------------------
// Sizes
#define BSZ 32
#define DIN 768
#define DT  384
#define HH  24
#define HW  576   // 24*24
#define NNODE 16

typedef _Float16 half_t;
typedef __attribute__((ext_vector_type(16))) _Float16 v16h;
typedef __attribute__((ext_vector_type(8)))  _Float16 v8h;
typedef __attribute__((ext_vector_type(8)))  float    v8f;
typedef unsigned int v4u __attribute__((ext_vector_type(4)));
typedef int v8i __attribute__((ext_vector_type(8)));
typedef int v4i __attribute__((ext_vector_type(4)));

#ifndef __has_builtin
#define __has_builtin(x) 0
#endif
#if __has_builtin(__builtin_amdgcn_tensor_load_to_lds) && \
    __has_builtin(__builtin_amdgcn_s_wait_tensorcnt)
#define HAVE_TDM 1
#else
#define HAVE_TDM 0
#endif

#define WMMA_F16(a, b, c) \
  __builtin_amdgcn_wmma_f32_16x16x32_f16(false, (a), false, (b), (short)0, (c), false, false)

// ---------------------------------------------------------------------------
// Fragment loaders (global).  A logical tile is 16x32 (MxK); B logical tile is
// 32x16 (KxN) stored as Bt[N][K] row-major (K contiguous).  Layouts used here (contiguous k order, not AMD's table order):
//   A: lane L -> row m = L%16, h = L/16; elems 0..15 = K[h*16 .. h*16+15]
//   B: lane L -> col n = L%16, h = L/16; elems 0..15 = K[h*16 .. h*16+15]
template <bool F32>
__device__ __forceinline__ v16h load_afrag(const void* Pv, int ld, int mb, int kk, int lane) {
  int m = mb + (lane & 15);
  int hf = lane >> 4;
  if constexpr (!F32) {
    const half_t* p = (const half_t*)Pv + (size_t)m * ld + kk + hf * 16;
    union { v16h v; v8h h[2]; } u;
    u.h[0] = *(const v8h*)p;
    u.h[1] = *(const v8h*)(p + 8);
    return u.v;
  } else {
    const float* p = (const float*)Pv + (size_t)m * ld + kk;
    v16h r;
#pragma unroll
    for (int e = 0; e < 8; e++) r[e] = (half_t)p[hf * 16 + e];
#pragma unroll
    for (int e = 0; e < 8; e++) r[8 + e] = (half_t)p[hf * 16 + 8 + e];
    return r;
  }
}

template <bool F32>
__device__ __forceinline__ v16h load_bfrag(const void* Pv, int ld, int nb, int kk, int lane) {
  int n = nb + (lane & 15);
  int hf = lane >> 4;
  if constexpr (!F32) {
    const half_t* p = (const half_t*)Pv + (size_t)n * ld + kk + hf * 16;
    union { v16h v; v8h h[2]; } u;
    u.h[0] = *(const v8h*)p;
    u.h[1] = *(const v8h*)(p + 8);
    return u.v;
  } else {
    const float* p = (const float*)Pv + (size_t)n * ld + kk + hf * 16;
    v16h r;
#pragma unroll
    for (int e = 0; e < 16; e++) r[e] = (half_t)p[e];
    return r;
  }
}

// LDS fragment loaders (panel layout: [rows][32] f16, K contiguous)
__device__ __forceinline__ v16h lds_afrag(const half_t* base, int ml, int lane) {
  int m = ml + (lane & 15);
  int hf = lane >> 4;
  const half_t* p = base + m * 32 + hf * 16;
  union { v16h v; v8h h[2]; } u;
  u.h[0] = *(const v8h*)p;
  u.h[1] = *(const v8h*)(p + 8);
  return u.v;
}
__device__ __forceinline__ v16h lds_bfrag(const half_t* base, int nl, int lane) {
  int n = nl + (lane & 15);
  int hf = lane >> 4;
  const half_t* p = base + n * 32 + hf * 16;
  union { v16h v; v8h h[2]; } u;
  u.h[0] = *(const v8h*)p;
  u.h[1] = *(const v8h*)(p + 8);
  return u.v;
}

#if HAVE_TDM
// Tensor Data Mover: 2D f16 tile (rows x 32 elems, global row stride ld elems)
// into LDS at byte offset ldsOff, packed [rows][32].  D# per CDNA5 ISA ch.8.
__device__ __forceinline__ void tdm_load_tile_f16(unsigned ldsOff, const half_t* g,
                                                  int rows, int ld) {
  unsigned long long ga = (unsigned long long)(size_t)g;
  unsigned long long st = (unsigned long long)(unsigned)ld;
  v4u g0;
  g0[0] = 1u;                                           // count=1, user mode
  g0[1] = ldsOff;                                       // lds_addr (bytes)
  g0[2] = (unsigned)(ga & 0xffffffffu);                 // global_addr[31:0]
  g0[3] = (unsigned)((ga >> 32) & 0x01ffffffu) | 0x80000000u;  // [56:32], type=2
  v8i g1;
  g1[0] = (int)(1u << 16);                              // data_size=1 (2 bytes)
  g1[1] = (int)(32u << 16);                             // tensor_dim0 lo16 (=32)
  g1[2] = (int)(((unsigned)rows & 0xffffu) << 16);      // dim0 hi=0 | tensor_dim1 lo
  g1[3] = (int)(32u << 16);                             // tensor_dim1 hi=0 | tile_dim0=32
  g1[4] = (int)((unsigned)rows & 0xffffu);              // tile_dim1 | tile_dim2=0
  g1[5] = (int)(st & 0xffffffffu);                      // tensor_dim0_stride[31:0]
  g1[6] = (int)((st >> 32) & 0xffffu);                  // stride[47:32] | dim1_stride lo=0
  g1[7] = 0;
  v4i z4 = {0, 0, 0, 0};
#if defined(__clang_major__) && __clang_major__ >= 23
  v8i z8 = {0, 0, 0, 0, 0, 0, 0, 0};
  __builtin_amdgcn_tensor_load_to_lds(g0, g1, z4, z4, z8, 0);
#else
  __builtin_amdgcn_tensor_load_to_lds(g0, g1, z4, z4, 0);
#endif
}
#else
// Fallback: cooperative global->LDS staging (dword copies).
__device__ __forceinline__ void stage_panel(half_t* dst, const half_t* src,
                                            int rows, int ld, int tid) {
  int dw = rows * 16;  // 32 f16 = 16 dwords per row
  for (int i = tid; i < dw; i += 256) {
    int r = i >> 4, c = i & 15;
    ((unsigned*)dst)[i] = ((const unsigned*)(src + (size_t)r * ld))[c];
  }
}
#endif

// Epilogue store: lane L holds column n = n0 + L%16, rows m0 + (L/16)*8 + v.
__device__ __forceinline__ void epi_store(
    v8f acc, int m0, int n0, float* C, int ldc, half_t* Ch, int ldch,
    const float* biasN, const float* biasM, const float* Sv,
    const float* Res, int flags, int lane) {
  int n  = n0 + (lane & 15);
  int mb = m0 + (lane >> 4) * 8;
  float bn = biasN ? biasN[n] : 0.f;
  float sv = (biasM && Sv) ? Sv[n] : 0.f;
#pragma unroll
  for (int v = 0; v < 8; v++) {
    int m = mb + v;
    float x = acc[v] + bn;
    if (biasM) x += biasM[m] * sv;
    if (flags & 1) x = fmaxf(x, 0.f);
    if (Res) x += Res[(size_t)m * ldc + n];
    if (C)  C[(size_t)m * ldc + n] = x;
    if (Ch) Ch[(size_t)m * ldch + n] = (half_t)x;
  }
}

// ---------------------------------------------------------------------------
// Generic batched NT GEMM:  C[b][m][n] = sum_k A[b][m][k] * B[b][n][k]
// A may be split along K (concat trick): k < kSplit -> A, else A2 (k-kSplit).
// 8 waves/block; wave computes 32x32 C; block tile 64(M) x 128(N).
// UT=true: Tensor-Data-Mover double-buffered LDS staging (f16 operands only).
template <bool AF32, bool BF32, bool UT>
__global__ void __launch_bounds__(256)
gemm_nt(const void* __restrict__ Av, const void* __restrict__ A2v, int kSplit,
        long sA, int lda,
        const void* __restrict__ Bv, long sB, int ldb,
        float* __restrict__ C, long sC, int ldc,
        half_t* __restrict__ Ch, long sCh, int ldch,
        const float* __restrict__ biasN,
        const float* __restrict__ biasM, const float* __restrict__ Svec, int svB,
        const float* __restrict__ Res, long sRes,
        int M, int N, int K, int flags) {
  int b    = blockIdx.z;
  int lane = threadIdx.x & 31;
  int wid  = threadIdx.x >> 5;
  int m0Blk = blockIdx.x * 64;
  int n0Blk = blockIdx.y * 128;
  int wmRaw = m0Blk + (wid >> 2) * 32;
  int wnRaw = n0Blk + (wid & 3) * 32;
  // Per-16 tile validity; clamp invalid tile bases to a valid in-panel address
  // so every wave loads/computes unconditionally (barrier-safe, no exec churn).
  bool mv0 = wmRaw < M,       mv1 = (wmRaw + 16) < M;
  bool nv0 = wnRaw < N,       nv1 = (wnRaw + 16) < N;
  int wm  = mv0 ? wmRaw      : m0Blk;
  int wm1 = mv1 ? wmRaw + 16 : wm;
  int wn  = nv0 ? wnRaw      : n0Blk;
  int wn1 = nv1 ? wnRaw + 16 : wn;

  const void* A  = AF32 ? (const void*)((const float*)Av  + (size_t)b * sA)
                        : (const void*)((const half_t*)Av + (size_t)b * sA);
  const void* A2 = AF32 ? (const void*)((const float*)A2v  + (size_t)b * sA)
                        : (const void*)((const half_t*)A2v + (size_t)b * sA);
  const void* B  = BF32 ? (const void*)((const float*)Bv  + (size_t)b * sB)
                        : (const void*)((const half_t*)Bv + (size_t)b * sB);
  if (C)  C  += (size_t)b * sC;
  if (Ch) Ch += (size_t)b * sCh;
  const float* Rp = Res  ? Res  + (size_t)b * sRes : nullptr;
  const float* Sv = Svec ? Svec + (size_t)b * svB  : nullptr;

  v8f c00 = {}; v8f c01 = {}; v8f c10 = {}; v8f c11 = {};

  if constexpr (UT) {
    __shared__ half_t lA[2][64 * 32];
    __shared__ half_t lB[2][128 * 32];
    int mRows = M - m0Blk; if (mRows > 64)  mRows = 64;
    int nRows = N - n0Blk; if (nRows > 128) nRows = 128;
    int ml0 = wm - m0Blk, ml1 = wm1 - m0Blk;
    int nl0 = wn - n0Blk, nl1 = wn1 - n0Blk;
    int nsteps = K >> 5;
#if HAVE_TDM
    if (wid == 0) {
      const half_t* Ap = (const half_t*)((0 < kSplit) ? A : A2);
      tdm_load_tile_f16((unsigned)(size_t)&lA[0][0],
                        Ap + (size_t)m0Blk * lda, mRows, lda);
      tdm_load_tile_f16((unsigned)(size_t)&lB[0][0],
                        (const half_t*)B + (size_t)n0Blk * ldb, nRows, ldb);
    }
#endif
    for (int s = 0; s < nsteps; s++) {
#if HAVE_TDM
      if (wid == 0) {
        if (s + 1 < nsteps) {
          int kn = (s + 1) << 5;
          const half_t* Ap; int k0;
          if (kn < kSplit) { Ap = (const half_t*)A;  k0 = kn; }
          else             { Ap = (const half_t*)A2; k0 = kn - kSplit; }
          int bn = (s + 1) & 1;
          tdm_load_tile_f16((unsigned)(size_t)&lA[bn][0],
                            Ap + (size_t)m0Blk * lda + k0, mRows, lda);
          tdm_load_tile_f16((unsigned)(size_t)&lB[bn][0],
                            (const half_t*)B + (size_t)n0Blk * ldb + kn, nRows, ldb);
          __builtin_amdgcn_s_wait_tensorcnt(2);  // pair for step s complete
        } else {
          __builtin_amdgcn_s_wait_tensorcnt(0);
        }
      }
      int bf = s & 1;
#else
      {
        int kk = s << 5;
        const half_t* Ap; int k0;
        if (kk < kSplit) { Ap = (const half_t*)A;  k0 = kk; }
        else             { Ap = (const half_t*)A2; k0 = kk - kSplit; }
        stage_panel(&lA[0][0], Ap + (size_t)m0Blk * lda + k0, mRows, lda, threadIdx.x);
        stage_panel(&lB[0][0], (const half_t*)B + (size_t)n0Blk * ldb + kk, nRows, ldb,
                    threadIdx.x);
      }
      int bf = 0;
#endif
      __syncthreads();
      v16h a0 = lds_afrag(&lA[bf][0], ml0, lane);
      v16h a1 = lds_afrag(&lA[bf][0], ml1, lane);
      v16h b0 = lds_bfrag(&lB[bf][0], nl0, lane);
      v16h b1 = lds_bfrag(&lB[bf][0], nl1, lane);
      c00 = WMMA_F16(a0, b0, c00);
      c01 = WMMA_F16(a0, b1, c01);
      c10 = WMMA_F16(a1, b0, c10);
      c11 = WMMA_F16(a1, b1, c11);
      __syncthreads();
    }
  } else {
    for (int kk = 0; kk < K; kk += 32) {
      const void* Ap = A; int k0 = kk;
      if (kk >= kSplit) { Ap = A2; k0 = kk - kSplit; }
      v16h a0 = load_afrag<AF32>(Ap, lda, wm,  k0, lane);
      v16h a1 = load_afrag<AF32>(Ap, lda, wm1, k0, lane);
      v16h b0 = load_bfrag<BF32>(B, ldb, wn,  kk, lane);
      v16h b1 = load_bfrag<BF32>(B, ldb, wn1, kk, lane);
      c00 = WMMA_F16(a0, b0, c00);
      c01 = WMMA_F16(a0, b1, c01);
      c10 = WMMA_F16(a1, b0, c10);
      c11 = WMMA_F16(a1, b1, c11);
    }
  }

  if (mv0 && nv0) epi_store(c00, wm,  wn,  C, ldc, Ch, ldch, biasN, biasM, Sv, Rp, flags, lane);
  if (mv0 && nv1) epi_store(c01, wm,  wn1, C, ldc, Ch, ldch, biasN, biasM, Sv, Rp, flags, lane);
  if (mv1 && nv0) epi_store(c10, wm1, wn,  C, ldc, Ch, ldch, biasN, biasM, Sv, Rp, flags, lane);
  if (mv1 && nv1) epi_store(c11, wm1, wn1, C, ldc, Ch, ldch, biasN, biasM, Sv, Rp, flags, lane);
}

// ---------------------------------------------------------------------------
// LayerNorm over channels; emits pixel-major f16 [b][p][c] (WMMA A-side) and
// channel-major f32 [b][c][p] (residual + f32 B-side) via LDS transpose.
__global__ void __launch_bounds__(256)
ln_kernel(const float* __restrict__ tok, const float* __restrict__ g,
          const float* __restrict__ be, half_t* __restrict__ fpx,
          float* __restrict__ fch) {
  __shared__ float smem[16 * 772];
  int b    = blockIdx.x;
  int pb   = blockIdx.y * 16;
  int lane = threadIdx.x & 31;
  int wid  = threadIdx.x >> 5;
#pragma unroll
  for (int q = 0; q < 2; q++) {
    int pp = wid * 2 + q;
    int p  = pb + pp;
    const float* row = tok + ((size_t)b * (HW + 2) + (p + 2)) * DIN;
    float xv[24];
    float s = 0.f;
#pragma unroll
    for (int k = 0; k < 24; k++) { xv[k] = row[lane + 32 * k]; s += xv[k]; }
#pragma unroll
    for (int d = 16; d >= 1; d >>= 1) s += __shfl_xor(s, d, 32);
    float mu = s * (1.f / 768.f);
    float vs = 0.f;
#pragma unroll
    for (int k = 0; k < 24; k++) { float d = xv[k] - mu; vs += d * d; }
#pragma unroll
    for (int d = 16; d >= 1; d >>= 1) vs += __shfl_xor(vs, d, 32);
    float rs = rsqrtf(vs * (1.f / 768.f) + 1e-5f);
#pragma unroll
    for (int k = 0; k < 24; k++) {
      int c = lane + 32 * k;
      float y = (xv[k] - mu) * rs * g[c] + be[c];
      smem[pp * 772 + c] = y;
      fpx[((size_t)b * HW + p) * DIN + c] = (half_t)y;
    }
  }
  __syncthreads();
  for (int idx = threadIdx.x; idx < 16 * DIN; idx += 256) {
    int c = idx >> 4, pp = idx & 15;
    fch[((size_t)b * DIN + c) * HW + pb + pp] = smem[pp * 772 + c];
  }
}

// Channel softmax of fc, keep channel 1 -> att[b*HW+p]. One wave per pixel.
__global__ void __launch_bounds__(256)
att_kernel(const float* __restrict__ fc, float* __restrict__ att) {
  int lane = threadIdx.x & 31;
  int row  = blockIdx.x * 8 + (threadIdx.x >> 5);
  const float* r = fc + (size_t)row * DT;
  float v[12], mx = -1e30f;
#pragma unroll
  for (int k = 0; k < 12; k++) { v[k] = r[lane + 32 * k]; mx = fmaxf(mx, v[k]); }
#pragma unroll
  for (int d = 16; d >= 1; d >>= 1) mx = fmaxf(mx, __shfl_xor(mx, d, 32));
  float s = 0.f;
#pragma unroll
  for (int k = 0; k < 12; k++) s += expf(v[k] - mx);
#pragma unroll
  for (int d = 16; d >= 1; d >>= 1) s += __shfl_xor(s, d, 32);
  if (lane == 0) att[row] = expf(r[1] - mx) / s;
}

// 4x4 mean pool of K*att, center-cropped 4x4 node grid -> V[b][n][d] (f16).
__global__ void __launch_bounds__(384)
pool_kernel(const float* __restrict__ Kf, const float* __restrict__ att,
            half_t* __restrict__ V) {
  int d = threadIdx.x;
  int n = blockIdx.x;
  int b = blockIdx.y;
  int i = n / 4 + 1, j = n % 4 + 1;
  float s = 0.f;
#pragma unroll
  for (int dy = 0; dy < 4; dy++)
#pragma unroll
    for (int dx = 0; dx < 4; dx++) {
      int p = (i * 4 + dy) * HH + (j * 4 + dx);
      s += Kf[((size_t)b * HW + p) * DT + d] * att[(size_t)b * HW + p];
    }
  V[((size_t)b * NNODE + n) * DT + d] = (half_t)(s * (1.f / 16.f));
}

// Softmax over nodes per pixel; emit [p][32]-padded and [n][p] f16 layouts plus
// S[b][n] = sum_p proj[n][p] (atomic; S pre-zeroed each launch).
__global__ void __launch_bounds__(256)
projsm_kernel(const float* __restrict__ p0, half_t* __restrict__ ppn,
              half_t* __restrict__ pnp, float* __restrict__ S) {
  int idx = blockIdx.x * 256 + threadIdx.x;
  if (idx >= BSZ * HW) return;
  int b = idx / HW, p = idx % HW;
  const float* r = p0 + (size_t)idx * NNODE;
  float v[NNODE], mx = -1e30f;
#pragma unroll
  for (int n = 0; n < NNODE; n++) { v[n] = r[n]; mx = fmaxf(mx, v[n]); }
  float s = 0.f;
#pragma unroll
  for (int n = 0; n < NNODE; n++) { v[n] = expf(v[n] - mx); s += v[n]; }
  float inv = 1.f / s;
#pragma unroll
  for (int n = 0; n < NNODE; n++) {
    float e = v[n] * inv;
    ppn[(size_t)idx * 32 + n] = (half_t)e;
    pnp[((size_t)b * NNODE + n) * HW + p] = (half_t)e;
    atomicAdd(&S[b * NNODE + n], e);
  }
}

// GCN node mixing: h[d][n] = sum_m Wg1[n][m]*ns[d][m] + bg1[n] - ns[d][n];
// write transposed f16 hT[b][n][d].
__global__ void __launch_bounds__(256)
gcn_kernel(const float* __restrict__ ns, const float* __restrict__ Wg1,
           const float* __restrict__ bg1, half_t* __restrict__ hT) {
  __shared__ float lns[DT * NNODE];
  int b = blockIdx.x;
  for (int idx = threadIdx.x; idx < DT * NNODE; idx += 256)
    lns[idx] = ns[(size_t)b * DT * NNODE + idx];
  __syncthreads();
  for (int d = threadIdx.x; d < DT; d += 256) {
    float nsd[NNODE];
#pragma unroll
    for (int m = 0; m < NNODE; m++) nsd[m] = lns[d * NNODE + m];
#pragma unroll
    for (int n = 0; n < NNODE; n++) {
      float h = bg1[n] - nsd[n];
#pragma unroll
      for (int m = 0; m < NNODE; m++) h += Wg1[n * NNODE + m] * nsd[m];
      hT[((size_t)b * NNODE + n) * DT + d] = (half_t)h;
    }
  }
}

__global__ void __launch_bounds__(256)
f2h_kernel(const float* __restrict__ s, half_t* __restrict__ d, int n) {
  int i = blockIdx.x * 256 + threadIdx.x;
  if (i < n) d[i] = (half_t)s[i];
}

// ---------------------------------------------------------------------------
extern "C" void kernel_launch(void* const* d_in, const int* in_sizes, int n_in,
                              void* d_out, int out_size, void* d_ws, size_t ws_size,
                              hipStream_t stream) {
  (void)in_sizes; (void)n_in; (void)out_size; (void)ws_size;
  const float* tok[2]  = {(const float*)d_in[0],  (const float*)d_in[1]};
  const float* lng[2]  = {(const float*)d_in[2],  (const float*)d_in[4]};
  const float* lnb[2]  = {(const float*)d_in[3],  (const float*)d_in[5]};
  const float* Wfc     = (const float*)d_in[6];
  const float* bfc     = (const float*)d_in[7];
  const float* Wq[2]   = {(const float*)d_in[8],  (const float*)d_in[16]};
  const float* bq[2]   = {(const float*)d_in[9],  (const float*)d_in[17]};
  const float* Wk[2]   = {(const float*)d_in[10], (const float*)d_in[18]};
  const float* bk[2]   = {(const float*)d_in[11], (const float*)d_in[19]};
  const float* Wext[2] = {(const float*)d_in[12], (const float*)d_in[20]};
  const float* Wg1[2]  = {(const float*)d_in[13], (const float*)d_in[21]};
  const float* bg1[2]  = {(const float*)d_in[14], (const float*)d_in[22]};
  const float* Wg2[2]  = {(const float*)d_in[15], (const float*)d_in[23]};
  float* outp = (float*)d_out;

  char* base = (char*)d_ws;
  size_t off = 0;
  auto alloc = [&](size_t bytes) -> char* {
    char* p = base + off;
    off += (bytes + 255) & ~(size_t)255;
    return p;
  };

  half_t* fpx[2]; float* fch[2]; float* Kf[2];
  for (int t = 0; t < 2; t++) fpx[t] = (half_t*)alloc((size_t)BSZ * HW * DIN * 2);
  for (int t = 0; t < 2; t++) fch[t] = (float*)alloc((size_t)BSZ * DIN * HW * 4);
  for (int t = 0; t < 2; t++) Kf[t]  = (float*)alloc((size_t)BSZ * HW * DT * 4);
  float* fcb  = (float*)alloc((size_t)BSZ * HW * DT * 4);
  float* attb = (float*)alloc((size_t)BSZ * HW * 4);

  char* zstart = base + off;
  half_t* ppn[2]; float* Sb[2]; half_t* M2[2];
  for (int t = 0; t < 2; t++) ppn[t] = (half_t*)alloc((size_t)BSZ * HW * 32 * 2);
  for (int t = 0; t < 2; t++) Sb[t]  = (float*)alloc((size_t)BSZ * NNODE * 4);
  for (int t = 0; t < 2; t++) M2[t]  = (half_t*)alloc((size_t)BSZ * DIN * 32 * 2);
  char* zend = base + off;

  float* p0[2]; half_t* pnp[2]; half_t* Vt[2]; half_t* Gt[2];
  float* nsb[2]; half_t* hT[2]; half_t* nrel[2];
  for (int t = 0; t < 2; t++) p0[t]   = (float*)alloc((size_t)BSZ * HW * NNODE * 4);
  for (int t = 0; t < 2; t++) pnp[t]  = (half_t*)alloc((size_t)BSZ * NNODE * HW * 2);
  for (int t = 0; t < 2; t++) Vt[t]   = (half_t*)alloc((size_t)BSZ * NNODE * DT * 2);
  for (int t = 0; t < 2; t++) Gt[t]   = (half_t*)alloc((size_t)BSZ * NNODE * DIN * 2);
  for (int t = 0; t < 2; t++) nsb[t]  = (float*)alloc((size_t)BSZ * DT * NNODE * 4);
  for (int t = 0; t < 2; t++) hT[t]   = (half_t*)alloc((size_t)BSZ * NNODE * DT * 2);
  for (int t = 0; t < 2; t++) nrel[t] = (half_t*)alloc((size_t)BSZ * NNODE * DT * 2);

  half_t* Wfc16 = (half_t*)alloc((size_t)DT * 2 * DIN * 2);
  half_t* Wq16[2]; half_t* Wk16[2]; half_t* We16[2]; half_t* Wg216[2];
  for (int t = 0; t < 2; t++) Wq16[t]  = (half_t*)alloc((size_t)DT * DIN * 2);
  for (int t = 0; t < 2; t++) Wk16[t]  = (half_t*)alloc((size_t)DT * DIN * 2);
  for (int t = 0; t < 2; t++) We16[t]  = (half_t*)alloc((size_t)DIN * DT * 2);
  for (int t = 0; t < 2; t++) Wg216[t] = (half_t*)alloc((size_t)DT * DT * 2);

  hipMemsetAsync(zstart, 0, (size_t)(zend - zstart), stream);

  auto cvt = [&](const float* s, half_t* d, int n) {
    f2h_kernel<<<(n + 255) / 256, 256, 0, stream>>>(s, d, n);
  };
  cvt(Wfc, Wfc16, DT * 2 * DIN);
  for (int t = 0; t < 2; t++) {
    cvt(Wq[t], Wq16[t], DT * DIN);
    cvt(Wk[t], Wk16[t], DT * DIN);
    cvt(Wext[t], We16[t], DIN * DT);
    cvt(Wg2[t], Wg216[t], DT * DT);
  }

  for (int t = 0; t < 2; t++)
    ln_kernel<<<dim3(BSZ, HW / 16), 256, 0, stream>>>(tok[t], lng[t], lnb[t],
                                                      fpx[t], fch[t]);

  const int BIG = 1 << 30;
  // G1: fc[p][o] = [f1px|f2px][p][c] x Wfc[o][c]  (M=576,N=384,K=1536) + bfc
  gemm_nt<false, false, true><<<dim3(9, 3, BSZ), 256, 0, stream>>>(
      fpx[0], fpx[1], DIN, (long)HW * DIN, DIN,
      Wfc16, 0, 2 * DIN,
      fcb, (long)HW * DT, DT, nullptr, 0, 0,
      bfc, nullptr, nullptr, 0, nullptr, 0,
      HW, DT, 2 * DIN, 0);

  att_kernel<<<(BSZ * HW) / 8, 256, 0, stream>>>(fcb, attb);

  for (int t = 0; t < 2; t++) {
    // G2: K[p][d] = fpx[p][c] x Wk[d][c] + bk   (M=576,N=384,K=768)
    gemm_nt<false, false, true><<<dim3(9, 3, BSZ), 256, 0, stream>>>(
        fpx[t], fpx[t], BIG, (long)HW * DIN, DIN,
        Wk16[t], 0, DIN,
        Kf[t], (long)HW * DT, DT, nullptr, 0, 0,
        bk[t], nullptr, nullptr, 0, nullptr, 0,
        HW, DT, DIN, 0);

    pool_kernel<<<dim3(NNODE, BSZ), 384, 0, stream>>>(Kf[t], attb, Vt[t]);

    // G3: proj0[p][n] = K[p][d] x V[n][d]   (M=576,N=16,K=384), A is f32
    gemm_nt<true, false, false><<<dim3(9, 1, BSZ), 256, 0, stream>>>(
        Kf[t], Kf[t], BIG, (long)HW * DT, DT,
        Vt[t], (long)NNODE * DT, DT,
        p0[t], (long)HW * NNODE, NNODE, nullptr, 0, 0,
        nullptr, nullptr, nullptr, 0, nullptr, 0,
        HW, NNODE, DT, 0);

    projsm_kernel<<<(BSZ * HW + 255) / 256, 256, 0, stream>>>(p0[t], ppn[t],
                                                              pnp[t], Sb[t]);

    // G5: Gt[n][c] = proj[n][p] x fch[c][p]  (M=16,N=768,K=576), B is f32
    gemm_nt<false, true, false><<<dim3(1, 6, BSZ), 256, 0, stream>>>(
        pnp[t], pnp[t], BIG, (long)NNODE * HW, HW,
        fch[t], (long)DIN * HW, HW,
        nullptr, 0, 0, Gt[t], (long)NNODE * DIN, DIN,
        nullptr, nullptr, nullptr, 0, nullptr, 0,
        NNODE, DIN, HW, 0);

    // G6: ns[d][n] = Wq[d][c] x Gt[n][c] + bq[d]*S[n]   (M=384,N=16,K=768)
    gemm_nt<false, false, false><<<dim3(6, 1, BSZ), 256, 0, stream>>>(
        Wq16[t], Wq16[t], BIG, 0, DIN,
        Gt[t], (long)NNODE * DIN, DIN,
        nsb[t], (long)DT * NNODE, NNODE, nullptr, 0, 0,
        nullptr, bq[t], Sb[t], NNODE, nullptr, 0,
        DT, NNODE, DIN, 0);

    gcn_kernel<<<BSZ, 256, 0, stream>>>(nsb[t], Wg1[t], bg1[t], hT[t]);

    // G7: nrelT[n][o] = relu( hT[n][d] x Wg2[o][d] )   (M=16,N=384,K=384)
    gemm_nt<false, false, false><<<dim3(1, 3, BSZ), 256, 0, stream>>>(
        hT[t], hT[t], BIG, (long)NNODE * DT, DT,
        Wg216[t], 0, DT,
        nullptr, 0, 0, nrel[t], (long)NNODE * DT, DT,
        nullptr, nullptr, nullptr, 0, nullptr, 0,
        NNODE, DT, DT, 1);

    // G8a: M2[c][n] = Wext[c][o] x nrelT[n][o]   (M=768,N=16,K=384)
    gemm_nt<false, false, false><<<dim3(12, 1, BSZ), 256, 0, stream>>>(
        We16[t], We16[t], BIG, 0, DT,
        nrel[t], (long)NNODE * DT, DT,
        nullptr, 0, 0, M2[t], (long)DIN * 32, 32,
        nullptr, nullptr, nullptr, 0, nullptr, 0,
        DIN, NNODE, DT, 0);

    // G8b: out[c][p] = fch[c][p] + M2[c][nk] x ppn[p][nk]  (M=768,N=576,K=32)
    gemm_nt<false, false, true><<<dim3(12, 5, BSZ), 256, 0, stream>>>(
        M2[t], M2[t], BIG, (long)DIN * 32, 32,
        ppn[t], (long)HW * 32, 32,
        outp + (size_t)t * BSZ * DIN * HW, (long)DIN * HW, HW,
        nullptr, 0, 0,
        nullptr, nullptr, nullptr, 0,
        fch[t], (long)DIN * HW,
        DIN, HW, 32, 0);
  }
}
